// TimeLSTM_v3_19387482374170
// MI455X (gfx1250) — compile-verified, matrix operand order checked by reading, not run on hardware
//
#include <hip/hip_runtime.h>

// TimeLSTM fused scan for gfx1250 (MI455X): one kernel runs all 512 steps.
// grid = B/16 blocks, block = 128 threads = 4 wave32 (one LSTM gate per wave).
// Recurrent + input GEMMs via v_wmma_f32_16x16x32_bf16; weights live in VGPRs.
// v2: software-pipelined x load, 2 barriers/step, branch-free activations,
//     cell state in registers.

#define B_SZ 2048
#define N_T  512
#define X_D  32
#define H_D  64

typedef __attribute__((ext_vector_type(16))) __bf16 v16bf;
typedef __attribute__((ext_vector_type(8)))  float  v8f;
typedef __attribute__((ext_vector_type(4)))  float  v4f;

// LDS row pads (elements): keep every vector-load base 32B aligned,
// and stagger banks across rows.
#define HP 80   // bf16 h rows: 160B stride
#define XP 48   // bf16 x rows: 96B stride
#define PP 72   // f32 preact rows

// Branch-free activations: v_exp_f32 + v_rcp_f32 (trans ops co-execute with WMMA).
__device__ __forceinline__ float fast_sigmoid(float v) {
  return __builtin_amdgcn_rcpf(1.0f + __expf(-v));
}
__device__ __forceinline__ float fast_tanh(float v) {
  return 1.0f - 2.0f * __builtin_amdgcn_rcpf(1.0f + __expf(2.0f * v));
}

__global__ __launch_bounds__(128)
void timelstm_fused_kernel(const float* __restrict__ x,
                           const float* __restrict__ ig_w_c, const float* __restrict__ ig_w_h,
                           const float* __restrict__ ig_w_x, const float* __restrict__ ig_b,
                           const float* __restrict__ fg_w_c, const float* __restrict__ fg_w_h,
                           const float* __restrict__ fg_w_x, const float* __restrict__ fg_b,
                           const float* __restrict__ in_w_h, const float* __restrict__ in_w_x,
                           const float* __restrict__ in_b,
                           const float* __restrict__ og_w_cn, const float* __restrict__ og_w_h,
                           const float* __restrict__ og_w_x, const float* __restrict__ og_b,
                           float* __restrict__ out)
{
  __shared__ alignas(32) __bf16 sh_h[16 * HP];     // h state, bf16, row-major
  __shared__ alignas(32) __bf16 sh_x[16 * XP];     // x_t staging, bf16
  __shared__ float sh_pre[4][16 * PP];             // per-gate preactivations

  const int tid  = threadIdx.x;
  const int lane = tid & 31;
  const int wave = tid >> 5;                       // gate id: 0=i 1=f 2=n 3=o
  const int b0   = blockIdx.x * 16;                // batch tile origin

  // ---- per-wave gate weight selection (wave-uniform) ----
  const float* Wh; const float* Wx; const float* bp;
  if (wave == 0)      { Wh = ig_w_h; Wx = ig_w_x; bp = ig_b; }
  else if (wave == 1) { Wh = fg_w_h; Wx = fg_w_x; bp = fg_b; }
  else if (wave == 2) { Wh = in_w_h; Wx = in_w_x; bp = in_b; }
  else                { Wh = og_w_h; Wx = og_w_x; bp = og_b; }

  const int nlo = lane & 15;            // column within 16-wide N tile
  const int kh  = (lane >> 4) << 4;     // 0 or 16: which K-half this lane holds

  // ---- preload WMMA B fragments in registers, f32 -> bf16 once ----
  // B[k,n] = W[n,k]  (out = act @ W^T). Lane n<16: K=k0..k0+15 of col n;
  // lane n+16: K=k0+16..k0+31 of col n  -> contiguous reads of W row n.
  v16bf bh[4][2];   // W_h: 4 N-tiles x 2 K-steps (K=64)
  v16bf bx[4];      // W_x: 4 N-tiles x 1 K-step  (K=32)
  #pragma unroll
  for (int nt = 0; nt < 4; ++nt) {
    #pragma unroll
    for (int kt = 0; kt < 2; ++kt) {
      const float* p = Wh + (nt * 16 + nlo) * H_D + kt * 32 + kh;
      v16bf r;
      #pragma unroll
      for (int i = 0; i < 16; ++i) r[i] = (__bf16)p[i];
      bh[nt][kt] = r;
    }
    const float* q = Wx + (nt * 16 + nlo) * X_D + kh;
    v16bf r;
    #pragma unroll
    for (int i = 0; i < 16; ++i) r[i] = (__bf16)q[i];
    bx[nt] = r;
  }
  float bias_r[4];
  #pragma unroll
  for (int nt = 0; nt < 4; ++nt) bias_r[nt] = bp[nt * 16 + nlo];

  // ---- elementwise phase mapping: thread -> (row em, 8 cols at ec) ----
  const int em = tid >> 3;
  const int ec = (tid & 7) * 8;
  float wci[8], wcf[8], wco[8];
  float cmr[8], hr[8];                 // cell + hidden state in registers
  #pragma unroll
  for (int i = 0; i < 8; ++i) {
    wci[i] = ig_w_c[ec + i];
    wcf[i] = fg_w_c[ec + i];
    wco[i] = og_w_cn[ec + i];
    cmr[i] = 0.0f;
    hr[i]  = 0.0f;
    sh_h[em * HP + ec + i] = (__bf16)0.0f;   // h_0 = 0 for first WMMA
  }

  // ---- software-pipelined x: hold x_t in regs, load x_{t+1} early ----
  const int xm = tid >> 3;             // staging row 0..15
  const int xc = (tid & 7) * 4;        // 4 floats per thread
  const float* xrow = x + (size_t)(b0 + xm) * N_T * X_D + xc;
  v4f xreg = *(const v4f*)(xrow);      // x_0

  for (int t = 0; t < N_T; ++t) {
    // ===== stage x_t (regs -> LDS bf16), then kick off x_{t+1} load =====
    {
      __bf16* d = sh_x + xm * XP + xc;
      d[0] = (__bf16)xreg.x; d[1] = (__bf16)xreg.y;
      d[2] = (__bf16)xreg.z; d[3] = (__bf16)xreg.w;
      const int tn = (t + 1 < N_T) ? (t + 1) : (N_T - 1);
      xreg = *(const v4f*)(xrow + (size_t)tn * X_D);        // waited next iter
      if (t + 2 < N_T) __builtin_prefetch(xrow + (size_t)(t + 2) * X_D, 0, 1);
    }
    __syncthreads();   // barrier 1: sh_x/sh_h ready for WMMA

    // ===== per-gate GEMM tile via WMMA =====
    // A as loaded here (contiguous k, same order as B): lane m<16 holds row m K=k0..k0+15; lane m+16 K=k0+16..k0+31.
    v16bf ax  = *(const v16bf*)(sh_x + nlo * XP + kh);       // x_t, K=0..31
    v16bf ah0 = *(const v16bf*)(sh_h + nlo * HP + 0  + kh);  // h,   K=0..31
    v16bf ah1 = *(const v16bf*)(sh_h + nlo * HP + 32 + kh);  // h,   K=32..63

    #pragma unroll
    for (int nt = 0; nt < 4; ++nt) {
      v8f acc;
      #pragma unroll
      for (int j = 0; j < 8; ++j) acc[j] = bias_r[nt];       // bias in C
      acc = __builtin_amdgcn_wmma_f32_16x16x32_bf16(false, ax,  false, bx[nt],
                                                    (short)0, acc, false, false);
      acc = __builtin_amdgcn_wmma_f32_16x16x32_bf16(false, ah0, false, bh[nt][0],
                                                    (short)0, acc, false, false);
      acc = __builtin_amdgcn_wmma_f32_16x16x32_bf16(false, ah1, false, bh[nt][1],
                                                    (short)0, acc, false, false);
      // C/D layout: VGPR j, lane<16 -> (M=j, N=lane); lane>=16 -> (M=8+j, N=lane-16)
      const int col   = nt * 16 + nlo;
      const int rbase = (lane >> 4) * 8;
      #pragma unroll
      for (int j = 0; j < 8; ++j)
        sh_pre[wave][(rbase + j) * PP + col] = acc[j];
    }
    __syncthreads();   // barrier 2: sh_pre ready; also fences sh_x reuse

    // ===== elementwise gate math, update cm (regs) and h (regs + LDS) =====
    #pragma unroll
    for (int i = 0; i < 8; ++i) {
      const int c = ec + i;
      const float pi = sh_pre[0][em * PP + c];
      const float pf = sh_pre[1][em * PP + c];
      const float pn = sh_pre[2][em * PP + c];
      const float po = sh_pre[3][em * PP + c];
      const float cm  = cmr[i];
      const float ig  = fast_sigmoid(wci[i] * cm + pi);
      const float fg  = fast_sigmoid(wcf[i] * cm + pf);
      const float nn  = fast_tanh(pn);
      const float cmn = fg * cm + ig * nn;
      const float og  = fast_sigmoid(wco[i] * cmn + po);
      const float hn  = og * fast_tanh(cmn);
      cmr[i] = cmn;
      hr[i]  = hn;
      sh_h[em * HP + c] = (__bf16)hn;
    }
    // no barrier here: barrier 1 of next iteration fences sh_h / sh_pre
  }

  // ---- final h -> out (f32, full precision from registers) ----
  #pragma unroll
  for (int i = 0; i < 8; ++i)
    out[(size_t)(b0 + em) * H_D + ec + i] = hr[i];
}

extern "C" void kernel_launch(void* const* d_in, const int* in_sizes, int n_in,
                              void* d_out, int out_size, void* d_ws, size_t ws_size,
                              hipStream_t stream) {
  (void)in_sizes; (void)n_in; (void)out_size; (void)d_ws; (void)ws_size;
  const float* x       = (const float*)d_in[0];
  const float* ig_w_c  = (const float*)d_in[1];
  const float* ig_w_h  = (const float*)d_in[2];
  const float* ig_w_x  = (const float*)d_in[3];
  const float* ig_b    = (const float*)d_in[4];
  const float* fg_w_c  = (const float*)d_in[5];
  const float* fg_w_h  = (const float*)d_in[6];
  const float* fg_w_x  = (const float*)d_in[7];
  const float* fg_b    = (const float*)d_in[8];
  const float* in_w_h  = (const float*)d_in[9];
  const float* in_w_x  = (const float*)d_in[10];
  const float* in_b    = (const float*)d_in[11];
  const float* og_w_cn = (const float*)d_in[12];
  const float* og_w_h  = (const float*)d_in[13];
  const float* og_w_x  = (const float*)d_in[14];
  const float* og_b    = (const float*)d_in[15];
  float* out = (float*)d_out;

  dim3 grid(B_SZ / 16);   // 128 blocks, one per 16-row batch tile
  dim3 block(128);        // 4 wave32: one gate per wave
  timelstm_fused_kernel<<<grid, block, 0, stream>>>(
      x, ig_w_c, ig_w_h, ig_w_x, ig_b,
      fg_w_c, fg_w_h, fg_w_x, fg_b,
      in_w_h, in_w_x, in_b,
      og_w_cn, og_w_h, og_w_x, og_b, out);
}
